// SSDGLNet_87617332838485
// MI455X (gfx1250) — matrix operand order repaired and checked by reading, not compiled, not run on hardware
//
#include <hip/hip_runtime.h>
#include <hip/hip_bf16.h>
#include <math.h>

typedef _Float16 f16;
typedef __attribute__((ext_vector_type(16))) _Float16 v16h;
typedef __attribute__((ext_vector_type(8)))  float    v8f;

#define TM 128
#define TN 64
#define TK 32
#define ASTRIDE 40   // f16 units, padded to dodge LDS bank conflicts
#define BTSTRIDE 40

union AFrag { v16h v; unsigned u[8]; };

#if defined(__has_builtin)
#if __has_builtin(__builtin_amdgcn_global_load_async_to_lds_b128) && \
    __has_builtin(__builtin_amdgcn_s_wait_asynccnt)
#define USE_ASYNC_LDS 1
#endif
#endif

#ifdef USE_ASYNC_LDS
typedef int v4i_ __attribute__((vector_size(16)));
typedef __attribute__((address_space(1))) v4i_ g_v4i;   // global
typedef __attribute__((address_space(3))) v4i_ l_v4i;   // LDS
#endif

// copy 16B global -> LDS; async (ASYNCcnt-tracked) when the toolchain has it
__device__ __forceinline__ void copyB16(const f16* __restrict__ src, f16* dst) {
#ifdef USE_ASYNC_LDS
  __builtin_amdgcn_global_load_async_to_lds_b128((g_v4i*)src, (l_v4i*)dst, 0, 0);
#else
  *(uint4*)dst = *(const uint4*)src;
#endif
}
__device__ __forceinline__ void asyncWait() {
#ifdef USE_ASYNC_LDS
  __builtin_amdgcn_s_wait_asynccnt(0);
#endif
}

__device__ __forceinline__ float sigmf(float x) { return 1.f / (1.f + expf(-x)); }

// =====================================================================
// Generic WMMA GEMM:  C[M,Nreal] = gatherA(M,K) * Bt(Npad rows, ldb=Kpad) [+bias]
//   mode 0: dense A0 (row stride lda0)
//   mode 1: concat [A0 (K0 cols, lda0) | A1 (lda1)]   (LSTM [x_t|h])
//   mode 2: implicit im2col of NHWC conv input
// B is N-major: Bt[n*ldb + k].  All M mult of 128, Npad mult of 64, Kpad mult of 32.
// =====================================================================
__global__ __launch_bounds__(256) void k_gemm(
    const float* __restrict__ A0, int lda0, int K0,
    const float* __restrict__ A1, int lda1,
    int mode,
    int convH, int convW, int Cin, int KH, int KW, int pad, int OW, int OHW,
    const f16* __restrict__ Bm, int ldb,
    float* __restrict__ C, int Nreal, int Kreal, int Kpad,
    const float* __restrict__ bias, int bias_mod)
{
  __shared__ __align__(16) f16 sA[TM * ASTRIDE];   // [row][k]
  __shared__ __align__(16) f16 sB[TN * BTSTRIDE];  // [col][k]
  const int tid  = threadIdx.x;
  const int M0   = blockIdx.x * TM;
  const int N0   = blockIdx.y * TN;
  const int lane = tid & 31;
  const int wave = tid >> 5;
  const int wm   = wave >> 1;      // 4 waves along M
  const int wn   = wave & 1;       // 2 waves along N
  const int half = lane >> 4;
  const int lr   = lane & 15;

  // ---- loop-invariant staging coordinates ----
  const int ar  = tid >> 1;            // A: 2 threads per row, 16 k each
  const int aks = (tid & 1) * 16;
  const int am  = M0 + ar;
  int cb = 0, coh = 0, cow = 0;
  if (mode == 2) { cb = am / OHW; const int r = am % OHW; coh = r / OW; cow = r % OW; }
  const float* aptr0 = A0 + (size_t)am * lda0 + aks;                    // modes 0/1
  const float* aptr1 = (mode == 1) ? (A1 + (size_t)am * lda1 + aks) : nullptr;
  // incremental decomposition of k = k0 + aks into (kh, kw, ci); valid when Cin%32==0
  const bool convfast = (mode == 2) && ((Cin & 31) == 0);
  int ci = aks, kwv = 0, khv = 0;
  const int bcol = tid >> 2;           // B: 4 threads per col, 8 k each
  const int bks  = (tid & 3) * 8;
  const f16* bsrc = Bm + (size_t)(N0 + bcol) * ldb + bks;
  f16* bdst = &sB[bcol * BTSTRIDE + bks];

  v8f acc[2][2] = {};

  for (int k0 = 0; k0 < Kpad; k0 += TK) {
    __syncthreads();
    // ---------------- A tile (128 x 32) ----------------
    {
      const int k = k0 + aks;
      float v[16];
      const float* src = nullptr;
      bool fast = false, zero = false;
      if (mode == 0) {
        if (k + 16 <= Kreal) { src = aptr0 + k0; fast = true; }
      } else if (mode == 1) {
        if (k + 16 <= K0) { src = aptr0 + k0; fast = true; }
        else if (k >= K0 && k + 16 <= Kreal) { src = aptr1 + (k0 - K0); fast = true; }
      } else if (convfast && k + 16 <= Kreal) {
        const int ih = coh + khv - pad, iw = cow + kwv - pad;
        fast = true;
        if ((unsigned)ih < (unsigned)convH && (unsigned)iw < (unsigned)convW)
          src = A0 + (((size_t)cb * convH + ih) * convW + iw) * Cin + ci;
        else
          zero = true;
      }
      if (fast) {
        if (!zero) {
          #pragma unroll
          for (int q = 0; q < 4; ++q) {
            const float4 f = ((const float4*)src)[q];
            v[q * 4 + 0] = f.x; v[q * 4 + 1] = f.y;
            v[q * 4 + 2] = f.z; v[q * 4 + 3] = f.w;
          }
        } else {
          #pragma unroll
          for (int i = 0; i < 16; ++i) v[i] = 0.f;
        }
      } else {
        // scalar fallback: enc1 (Cin=1, Kreal=9) and K tails; keep compact
        #pragma unroll 1
        for (int i = 0; i < 16; ++i) {
          const int kk = k + i;
          float x = 0.f;
          if (mode == 0) {
            if (kk < Kreal) x = A0[(size_t)am * lda0 + kk];
          } else if (mode == 1) {
            if (kk < K0)         x = A0[(size_t)am * lda0 + kk];
            else if (kk < Kreal) x = A1[(size_t)am * lda1 + (kk - K0)];
          } else if (kk < Kreal) {
            const int cci = kk % Cin, t = kk / Cin, kw = t % KW, kh = t / KW;
            const int ih = coh + kh - pad, iw = cow + kw - pad;
            if ((unsigned)ih < (unsigned)convH && (unsigned)iw < (unsigned)convW)
              x = A0[(((size_t)cb * convH + ih) * convW + iw) * Cin + cci];
          }
          v[i] = x;
        }
      }
      union { f16 h[16]; uint4 q[2]; } tv;
      #pragma unroll
      for (int i = 0; i < 16; ++i) tv.h[i] = (f16)v[i];
      *(uint4*)&sA[ar * ASTRIDE + aks]     = tv.q[0];
      *(uint4*)&sA[ar * ASTRIDE + aks + 8] = tv.q[1];
    }
    // advance incremental conv-k state by TK (at most one carry: Cin%32==0)
    if (convfast) {
      ci += TK;
      if (ci >= Cin) {
        ci -= Cin;
        if (++kwv == KW) { kwv = 0; ++khv; }
      }
    }
    // ---------------- B tile (64 cols x 32 k), contiguous along k ----------------
    copyB16(bsrc + k0, bdst);
    if (k0 + TK < Kpad) __builtin_prefetch(bsrc + k0 + TK, 0, 1);
    asyncWait();
    __syncthreads();

    // ---- per-lane fragment gather (ISA 16x32 A / 32x16 B layouts) ----
    AFrag af[2], bf[2];
    #pragma unroll
    for (int mi = 0; mi < 2; ++mi) {
      const int row = wm * 32 + mi * 16 + lr;
      #pragma unroll
      for (int q = 0; q < 8; ++q) {
        const int kk = (q < 4 ? q * 2 : 16 + (q - 4) * 2) + half * 8;
        af[mi].u[q] = *(const unsigned*)&sA[row * ASTRIDE + kk];
      }
    }
    #pragma unroll
    for (int ni = 0; ni < 2; ++ni) {
      const int col  = wn * 32 + ni * 16 + lr;
      const int base = col * BTSTRIDE;
      #pragma unroll
      for (int q = 0; q < 8; ++q) {
        const int kk = (q < 4 ? q * 2 : 16 + (q - 4) * 2) + half * 8;
        bf[ni].u[q] = *(const unsigned*)&sB[base + kk];
      }
    }
    #pragma unroll
    for (int mi = 0; mi < 2; ++mi)
      #pragma unroll
      for (int ni = 0; ni < 2; ++ni)
        acc[mi][ni] = __builtin_amdgcn_wmma_f32_16x16x32_f16(
            false, af[mi].v, false, bf[ni].v, (short)0, acc[mi][ni], false, false);
  }

  // ---- store C (8-VGPR f32 C/D layout) ----
  #pragma unroll
  for (int mi = 0; mi < 2; ++mi)
    #pragma unroll
    for (int ni = 0; ni < 2; ++ni) {
      const int colg = N0 + wn * 32 + ni * 16 + lr;
      if (colg < Nreal) {
        const float bv = bias ? bias[colg % bias_mod] : 0.f;
        #pragma unroll
        for (int r = 0; r < 8; ++r) {
          const int rowg = M0 + wm * 32 + mi * 16 + half * 8 + r;
          C[(size_t)rowg * Nreal + colg] = acc[mi][ni][r] + bv;
        }
      }
    }
}

// ============== weight transforms (fp32 -> f16, N-major Bt[n][k]) ==============
// conv w[Cout,Cin,KH,KW] -> Bt[co, (kh*KW+kw)*Cin+ci], zero-padded Npad x Kpad
__global__ void k_wt_conv(const float* __restrict__ w, f16* __restrict__ Bt,
                          int Cout, int Cin, int KH, int KW, int Kpad, int Npad) {
  const long long total = (long long)Npad * Kpad;
  for (long long idx = (long long)blockIdx.x * 256 + threadIdx.x; idx < total;
       idx += (long long)gridDim.x * 256) {
    const int n = (int)(idx / Kpad), k = (int)(idx % Kpad);
    float v = 0.f;
    if (n < Cout && k < Cin * KH * KW) {
      const int ci = k % Cin, t = k / Cin, kw = t % KW, kh = t / KW;
      v = w[(((size_t)n * Cin + ci) * KH + kh) * KW + kw];
    }
    Bt[idx] = (f16)v;
  }
}
// LSTM: Bt[j, k] = k<256 ? wih[j,k] : whh[j,k-256]   (N=1024 rows, K=512)
__global__ void k_wt_lstm(const float* __restrict__ wih, const float* __restrict__ whh,
                          f16* __restrict__ Bt) {
  const long long total = 1024LL * 512;
  for (long long idx = (long long)blockIdx.x * 256 + threadIdx.x; idx < total;
       idx += (long long)gridDim.x * 256) {
    const int n = (int)(idx >> 9), k = (int)(idx & 511);
    const float v = (k < 256) ? wih[(size_t)n * 256 + k] : whh[(size_t)n * 256 + (k - 256)];
    Bt[idx] = (f16)v;
  }
}
// convT2x2s2: w[Cin,Cout,2,2] -> Bt[(kh*2+kw)*Cout+co, ci]
__global__ void k_wt_deconv(const float* __restrict__ w, f16* __restrict__ Bt,
                            int Cin, int Cout) {
  const long long total = (long long)Cin * 4 * Cout;
  for (long long idx = (long long)blockIdx.x * 256 + threadIdx.x; idx < total;
       idx += (long long)gridDim.x * 256) {
    const int n = (int)(idx / Cin), k = (int)(idx % Cin);
    const int kk = n / Cout, co = n % Cout, kh = kk >> 1, kw = kk & 1;
    Bt[idx] = (f16)w[(((size_t)k * Cout + co) * 2 + kh) * 2 + kw];
  }
}

// ==================== BN (batch stats), ReLU, pool ====================
__global__ __launch_bounds__(256) void k_bnstats(const float* __restrict__ X,
    long long Mrows, int Ncols, int C, float* __restrict__ mean, float* __restrict__ rstd) {
  const int c = blockIdx.x;
  const int rep = Ncols / C;
  float s = 0.f, s2 = 0.f;
  for (long long row = threadIdx.x; row < Mrows; row += 256) {
    const float* base = X + row * Ncols + c;
    #pragma unroll 1
    for (int j = 0; j < rep; ++j) {
      const float v = base[(size_t)j * C];
      s += v; s2 += v * v;
    }
  }
  __shared__ float sh[256], sh2[256];
  sh[threadIdx.x] = s; sh2[threadIdx.x] = s2; __syncthreads();
  for (int o = 128; o > 0; o >>= 1) {
    if ((int)threadIdx.x < o) { sh[threadIdx.x] += sh[threadIdx.x + o]; sh2[threadIdx.x] += sh2[threadIdx.x + o]; }
    __syncthreads();
  }
  if (threadIdx.x == 0) {
    const float n = (float)(Mrows * rep);
    const float m = sh[0] / n;
    const float var = sh2[0] / n - m * m;
    mean[c] = m; rstd[c] = rsqrtf(var + 1e-5f);
  }
}
// mode 0: Y same layout as X. mode 1: deconv scatter [M,4C] -> NHWC [B,2Hi,2Wi,C]
__global__ void k_bnapply(const float* __restrict__ X, float* __restrict__ Y,
    const float* __restrict__ mean, const float* __restrict__ rstd,
    const float* __restrict__ g, const float* __restrict__ bb,
    long long total, int Ncols, int C, int mode, int Hi, int Wi) {
  for (long long idx = (long long)blockIdx.x * 256 + threadIdx.x; idx < total;
       idx += (long long)gridDim.x * 256) {
    const long long row = idx / Ncols; const int col = (int)(idx % Ncols);
    const int c = col % C;
    float v = (X[idx] - mean[c]) * rstd[c] * g[c] + bb[c];
    v = fmaxf(v, 0.f);
    if (mode == 0) { Y[idx] = v; }
    else {
      const long long b = row / (Hi * Wi); const int r = (int)(row % (Hi * Wi));
      const int hi = r / Wi, wi = r % Wi;
      const int kk = col / C, kh = kk >> 1, kw = kk & 1;
      Y[(((size_t)b * (2 * Hi) + (2 * hi + kh)) * (2 * Wi) + (2 * wi + kw)) * C + c] = v;
    }
  }
}
__global__ void k_maxpool(const float* __restrict__ X, float* __restrict__ Y,
                          int H, int W, int C, long long total) {
  const int HO = H / 2, WO = W / 2;
  for (long long idx = (long long)blockIdx.x * 256 + threadIdx.x; idx < total;
       idx += (long long)gridDim.x * 256) {
    const int c = (int)(idx % C); long long t = idx / C;
    const int wo = (int)(t % WO); t /= WO;
    const int ho = (int)(t % HO); const long long b = t / HO;
    const size_t base = (((size_t)b * H + 2 * ho) * W + 2 * wo) * C + c;
    float m = X[base];
    m = fmaxf(m, X[base + C]);
    m = fmaxf(m, X[base + (size_t)W * C]);
    m = fmaxf(m, X[base + (size_t)W * C + C]);
    Y[idx] = m;
  }
}
__global__ void k_zero(float* __restrict__ p, long long n) {
  for (long long i = (long long)blockIdx.x * 256 + threadIdx.x; i < n;
       i += (long long)gridDim.x * 256) p[i] = 0.f;
}

// ==================== LSTM gates ====================
__global__ void k_lstm_gate(const float* __restrict__ G,
    const float* __restrict__ bih, const float* __restrict__ bhh,
    const float* __restrict__ cin, float* __restrict__ hout, float* __restrict__ cout,
    float* __restrict__ outbuf, long long outstride) {
  const long long total = 2048LL * 256;
  for (long long idx = (long long)blockIdx.x * 256 + threadIdx.x; idx < total;
       idx += (long long)gridDim.x * 256) {
    const long long m = idx >> 8; const int j = (int)(idx & 255);
    const float* gr = G + m * 1024;
    const float ig = gr[j]       + bih[j]       + bhh[j];
    const float fg = gr[256 + j] + bih[256 + j] + bhh[256 + j];
    const float gg = gr[512 + j] + bih[512 + j] + bhh[512 + j];
    const float og = gr[768 + j] + bih[768 + j] + bhh[768 + j];
    const float c2 = sigmf(fg) * cin[idx] + sigmf(ig) * tanhf(gg);
    const float h2 = sigmf(og) * tanhf(c2);
    cout[idx] = c2; hout[idx] = h2;
    if (outbuf) outbuf[m * outstride + j] = h2;
  }
}

// ==================== GJAM attention ====================
__global__ __launch_bounds__(256) void k_ca(const float* __restrict__ X,
    const float* __restrict__ W1, const float* __restrict__ b1,
    const float* __restrict__ W2, const float* __restrict__ b2,
    float* __restrict__ Y) {
  const int b = blockIdx.x, c = threadIdx.x;
  __shared__ float avg[256], r[32];
  float s = 0.f;
  #pragma unroll
  for (int p = 0; p < 8; ++p) s += X[(size_t)b * 2048 + p * 256 + c];
  avg[c] = s * 0.125f;
  __syncthreads();
  if (c < 32) {
    float a = b1[c];
    for (int k = 0; k < 256; ++k) a += W1[(size_t)c * 256 + k] * avg[k];
    r[c] = fmaxf(a, 0.f);
  }
  __syncthreads();
  float a = b2[c];
  #pragma unroll
  for (int j = 0; j < 32; ++j) a += W2[(size_t)c * 32 + j] * r[j];
  const float ca = sigmf(a);
  #pragma unroll
  for (int p = 0; p < 8; ++p) {
    const size_t i = (size_t)b * 2048 + p * 256 + c;
    Y[i] = X[i] * ca;
  }
}
__global__ __launch_bounds__(256) void k_sa(const float* __restrict__ X,
    const float* __restrict__ Wsa, const float* __restrict__ bsa,
    float* __restrict__ Y) {
  const int b = blockIdx.x, c = threadIdx.x;
  __shared__ float smax[8], smean[8], ssig[8];
  if (c < 8) {
    float mx = -1e30f, sm = 0.f;
    for (int k = 0; k < 256; ++k) {
      const float v = X[(size_t)b * 2048 + c * 256 + k];
      mx = fmaxf(mx, v); sm += v;
    }
    smax[c] = mx; smean[c] = sm / 256.f;
  }
  __syncthreads();
  if (c < 8) {
    const int h = c >> 1, w = c & 1;
    float a = bsa[0];
    for (int kh = 0; kh < 7; ++kh) {
      const int ih = h + kh - 3;
      if (ih < 0 || ih >= 4) continue;
      for (int kw = 0; kw < 7; ++kw) {
        const int iw = w + kw - 3;
        if (iw < 0 || iw >= 2) continue;
        const int pos = ih * 2 + iw;
        a += smax[pos]  * Wsa[kh * 7 + kw];
        a += smean[pos] * Wsa[49 + kh * 7 + kw];
      }
    }
    ssig[c] = sigmf(a);
  }
  __syncthreads();
  #pragma unroll
  for (int p = 0; p < 8; ++p) {
    const size_t i = (size_t)b * 2048 + p * 256 + c;
    Y[i] = X[i] * ssig[p];
  }
}

// head output [B*512, 10] row-major -> NCHW [2048,10,32,16]
__global__ void k_final(const float* __restrict__ G, float* __restrict__ out, long long total) {
  for (long long idx = (long long)blockIdx.x * 256 + threadIdx.x; idx < total;
       idx += (long long)gridDim.x * 256) {
    const int w  = (int)(idx & 15);
    const int h  = (int)((idx >> 4) & 31);
    const int co = (int)((idx >> 9) % 10);
    const long long b = idx / 5120;
    out[idx] = G[((size_t)b * 512 + h * 16 + w) * 10 + co];
  }
}

static inline unsigned ewb(long long n) {
  long long g = (n + 255) / 256;
  return (unsigned)(g > 1048576 ? 1048576 : g);
}

extern "C" void kernel_launch(void* const* d_in, const int* in_sizes, int n_in,
                              void* d_out, int out_size, void* d_ws, size_t ws_size,
                              hipStream_t stream) {
  (void)in_sizes; (void)n_in; (void)out_size; (void)ws_size;
  const float* x      = (const float*)d_in[0];
  const float* enc_w1 = (const float*)d_in[1];
  const float* bn1_g  = (const float*)d_in[3];
  const float* bn1_b  = (const float*)d_in[4];
  const float* enc_w2 = (const float*)d_in[5];
  const float* bn2_g  = (const float*)d_in[7];
  const float* bn2_b  = (const float*)d_in[8];
  const float* enc_w3 = (const float*)d_in[9];
  const float* bn3_g  = (const float*)d_in[11];
  const float* bn3_b  = (const float*)d_in[12];
  const float* gcl_w  = (const float*)d_in[13];
  const float* gcl_bng = (const float*)d_in[15];
  const float* gcl_bnb = (const float*)d_in[16];
  const float* l0_wih = (const float*)d_in[17];
  const float* l0_whh = (const float*)d_in[18];
  const float* l0_bih = (const float*)d_in[19];
  const float* l0_bhh = (const float*)d_in[20];
  const float* l1_wih = (const float*)d_in[21];
  const float* l1_whh = (const float*)d_in[22];
  const float* l1_bih = (const float*)d_in[23];
  const float* l1_bhh = (const float*)d_in[24];
  const float* ca_w1  = (const float*)d_in[25];
  const float* ca_b1  = (const float*)d_in[26];
  const float* ca_w2  = (const float*)d_in[27];
  const float* ca_b2  = (const float*)d_in[28];
  const float* sa_w   = (const float*)d_in[29];
  const float* sa_b   = (const float*)d_in[30];
  const float* dec_w1 = (const float*)d_in[31];
  const float* dbn1_g = (const float*)d_in[33];
  const float* dbn1_b = (const float*)d_in[34];
  const float* dec_w2 = (const float*)d_in[35];
  const float* dbn2_g = (const float*)d_in[37];
  const float* dbn2_b = (const float*)d_in[38];
  const float* dec_w3 = (const float*)d_in[39];
  const float* dbn3_g = (const float*)d_in[41];
  const float* dbn3_b = (const float*)d_in[42];
  const float* out_w  = (const float*)d_in[43];
  const float* out_b  = (const float*)d_in[44];

  char* ws = (char*)d_ws;
  const size_t MB = 1ull << 20;
  float* gbuf  = (float*)(ws + 0);          // GEMM outputs (reused), 256 MiB
  float* abuf  = (float*)(ws + 256 * MB);   // full-res activations, 256 MiB
  float* pool1 = (float*)(ws + 512 * MB);   // [2048,16,8,64]
  float* pool2 = (float*)(ws + 576 * MB);   // [2048,8,4,128]
  float* pool3 = (float*)(ws + 608 * MB);   // [2048,4,2,256]
  float* ybuf  = (float*)(ws + 624 * MB);   // GCL out [2048,8,256]
  float* xlstm = (float*)(ws + 640 * MB);   // LSTM out NHWC [2048,8,256]
  float* xatt  = (float*)(ws + 656 * MB);   // attention-applied
  float* dbuf1 = (float*)(ws + 672 * MB);   // [2048,8,4,128]
  float* dbuf2 = (float*)(ws + 704 * MB);   // [2048,16,8,64]
  float* h0    = (float*)(ws + 768 * MB);   // 4 contiguous 2-MiB state bufs
  float* c0    = (float*)(ws + 770 * MB);
  float* h1    = (float*)(ws + 772 * MB);
  float* c1    = (float*)(ws + 774 * MB);
  float* gates = (float*)(ws + 776 * MB);   // [2048,1024]
  float* meanb = (float*)(ws + 784 * MB);
  float* rstdb = (float*)(ws + 784 * MB + 4096);
  f16* wt_e1 = (f16*)(ws + 785 * MB);
  f16* wt_e2 = (f16*)(ws + 786 * MB);
  f16* wt_e3 = (f16*)(ws + 787 * MB);
  f16* wt_gc = (f16*)(ws + 788 * MB);
  f16* wt_l0 = (f16*)(ws + 792 * MB);
  f16* wt_l1 = (f16*)(ws + 793 * MB);
  f16* wt_d1 = (f16*)(ws + 794 * MB);
  f16* wt_d2 = (f16*)(ws + 795 * MB);
  f16* wt_d3 = (f16*)(ws + 796 * MB);
  f16* wt_hd = (f16*)(ws + 797 * MB);

  // ---- weight transforms (fp32 -> f16 N-major B matrices) ----
  k_wt_conv<<<ewb(64 * 32), 256, 0, stream>>>(enc_w1, wt_e1, 64, 1, 3, 3, 32, 64);
  k_wt_conv<<<ewb(128 * 576), 256, 0, stream>>>(enc_w2, wt_e2, 128, 64, 3, 3, 576, 128);
  k_wt_conv<<<ewb(256 * 1152), 256, 0, stream>>>(enc_w3, wt_e3, 256, 128, 3, 3, 1152, 256);
  k_wt_conv<<<ewb(256 * 6400), 256, 0, stream>>>(gcl_w, wt_gc, 256, 256, 5, 5, 6400, 256);
  k_wt_conv<<<ewb(64 * 64), 256, 0, stream>>>(out_w, wt_hd, 10, 64, 1, 1, 64, 64);
  k_wt_lstm<<<ewb(1024 * 512), 256, 0, stream>>>(l0_wih, l0_whh, wt_l0);
  k_wt_lstm<<<ewb(1024 * 512), 256, 0, stream>>>(l1_wih, l1_whh, wt_l1);
  k_wt_deconv<<<ewb(256 * 512), 256, 0, stream>>>(dec_w1, wt_d1, 256, 128);
  k_wt_deconv<<<ewb(128 * 256), 256, 0, stream>>>(dec_w2, wt_d2, 128, 64);
  k_wt_deconv<<<ewb(64 * 256), 256, 0, stream>>>(dec_w3, wt_d3, 64, 64);

  // ---- encoder stage 1: conv3x3(1->64) + BN + ReLU + pool ----
  {
    dim3 g(1048576 / TM, 64 / TN);
    k_gemm<<<g, 256, 0, stream>>>(x, 0, 0, nullptr, 0, 2, 32, 16, 1, 3, 3, 1, 16, 512,
                                  wt_e1, 32, gbuf, 64, 9, 32, nullptr, 1);
    k_bnstats<<<64, 256, 0, stream>>>(gbuf, 1048576LL, 64, 64, meanb, rstdb);
    const long long tot = 1048576LL * 64;
    k_bnapply<<<ewb(tot), 256, 0, stream>>>(gbuf, abuf, meanb, rstdb, bn1_g, bn1_b, tot, 64, 64, 0, 0, 0);
    const long long pt = 2048LL * 16 * 8 * 64;
    k_maxpool<<<ewb(pt), 256, 0, stream>>>(abuf, pool1, 32, 16, 64, pt);
  }
  // ---- stage 2: conv3x3(64->128) ----
  {
    dim3 g(262144 / TM, 128 / TN);
    k_gemm<<<g, 256, 0, stream>>>(pool1, 0, 0, nullptr, 0, 2, 16, 8, 64, 3, 3, 1, 8, 128,
                                  wt_e2, 576, gbuf, 128, 576, 576, nullptr, 1);
    k_bnstats<<<128, 256, 0, stream>>>(gbuf, 262144LL, 128, 128, meanb, rstdb);
    const long long tot = 262144LL * 128;
    k_bnapply<<<ewb(tot), 256, 0, stream>>>(gbuf, abuf, meanb, rstdb, bn2_g, bn2_b, tot, 128, 128, 0, 0, 0);
    const long long pt = 2048LL * 8 * 4 * 128;
    k_maxpool<<<ewb(pt), 256, 0, stream>>>(abuf, pool2, 16, 8, 128, pt);
  }
  // ---- stage 3: conv3x3(128->256) ----
  {
    dim3 g(65536 / TM, 256 / TN);
    k_gemm<<<g, 256, 0, stream>>>(pool2, 0, 0, nullptr, 0, 2, 8, 4, 128, 3, 3, 1, 4, 32,
                                  wt_e3, 1152, gbuf, 256, 1152, 1152, nullptr, 1);
    k_bnstats<<<256, 256, 0, stream>>>(gbuf, 65536LL, 256, 256, meanb, rstdb);
    const long long tot = 65536LL * 256;
    k_bnapply<<<ewb(tot), 256, 0, stream>>>(gbuf, abuf, meanb, rstdb, bn3_g, bn3_b, tot, 256, 256, 0, 0, 0);
    const long long pt = 2048LL * 4 * 2 * 256;
    k_maxpool<<<ewb(pt), 256, 0, stream>>>(abuf, pool3, 8, 4, 256, pt);
  }
  // ---- GCL: conv5x5(256->256) pad2 + BN + ReLU ----
  {
    dim3 g(16384 / TM, 256 / TN);
    k_gemm<<<g, 256, 0, stream>>>(pool3, 0, 0, nullptr, 0, 2, 4, 2, 256, 5, 5, 2, 2, 8,
                                  wt_gc, 6400, gbuf, 256, 6400, 6400, nullptr, 1);
    k_bnstats<<<256, 256, 0, stream>>>(gbuf, 16384LL, 256, 256, meanb, rstdb);
    const long long tot = 16384LL * 256;
    k_bnapply<<<ewb(tot), 256, 0, stream>>>(gbuf, ybuf, meanb, rstdb, gcl_bng, gcl_bnb, tot, 256, 256, 0, 0, 0);
  }
  // ---- 2-layer LSTM, T=8 (concat-mode GEMM: [x_t|h] @ [Wih;Whh]^T) ----
  k_zero<<<ewb(4LL * 2048 * 256), 256, 0, stream>>>(h0, 4LL * 2048 * 256);
  for (int t = 0; t < 8; ++t) {
    dim3 g(2048 / TM, 1024 / TN);
    k_gemm<<<g, 256, 0, stream>>>(ybuf + t * 256, 2048, 256, h0, 256, 1,
                                  0, 0, 0, 0, 0, 0, 0, 0,
                                  wt_l0, 512, gates, 1024, 512, 512, nullptr, 1);
    k_lstm_gate<<<2048, 256, 0, stream>>>(gates, l0_bih, l0_bhh, c0, h0, c0, nullptr, 0);
    k_gemm<<<g, 256, 0, stream>>>(h0, 256, 256, h1, 256, 1,
                                  0, 0, 0, 0, 0, 0, 0, 0,
                                  wt_l1, 512, gates, 1024, 512, 512, nullptr, 1);
    k_lstm_gate<<<2048, 256, 0, stream>>>(gates, l1_bih, l1_bhh, c1, h1, c1, xlstm + t * 256, 2048);
  }
  // ---- GJAM: channel attention then spatial attention ----
  k_ca<<<2048, 256, 0, stream>>>(xlstm, ca_w1, ca_b1, ca_w2, ca_b2, xatt);
  k_sa<<<2048, 256, 0, stream>>>(xatt, sa_w, sa_b, xatt);
  // ---- decoder: 3x (convT2x2s2 as GEMM + BN-scatter + ReLU) ----
  {
    dim3 g(16384 / TM, 512 / TN);
    k_gemm<<<g, 256, 0, stream>>>(xatt, 256, 256, nullptr, 0, 0, 0, 0, 0, 0, 0, 0, 0, 0,
                                  wt_d1, 256, gbuf, 512, 256, 256, nullptr, 1);
    k_bnstats<<<128, 256, 0, stream>>>(gbuf, 16384LL, 512, 128, meanb, rstdb);
    const long long tot = 16384LL * 512;
    k_bnapply<<<ewb(tot), 256, 0, stream>>>(gbuf, dbuf1, meanb, rstdb, dbn1_g, dbn1_b, tot, 512, 128, 1, 4, 2);
  }
  {
    dim3 g(65536 / TM, 256 / TN);
    k_gemm<<<g, 256, 0, stream>>>(dbuf1, 128, 128, nullptr, 0, 0, 0, 0, 0, 0, 0, 0, 0, 0,
                                  wt_d2, 128, gbuf, 256, 128, 128, nullptr, 1);
    k_bnstats<<<64, 256, 0, stream>>>(gbuf, 65536LL, 256, 64, meanb, rstdb);
    const long long tot = 65536LL * 256;
    k_bnapply<<<ewb(tot), 256, 0, stream>>>(gbuf, dbuf2, meanb, rstdb, dbn2_g, dbn2_b, tot, 256, 64, 1, 8, 4);
  }
  {
    dim3 g(262144 / TM, 256 / TN);
    k_gemm<<<g, 256, 0, stream>>>(dbuf2, 64, 64, nullptr, 0, 0, 0, 0, 0, 0, 0, 0, 0, 0,
                                  wt_d3, 64, gbuf, 256, 64, 64, nullptr, 1);
    k_bnstats<<<64, 256, 0, stream>>>(gbuf, 262144LL, 256, 64, meanb, rstdb);
    const long long tot = 262144LL * 256;
    k_bnapply<<<ewb(tot), 256, 0, stream>>>(gbuf, abuf, meanb, rstdb, dbn3_g, dbn3_b, tot, 256, 64, 1, 16, 8);
  }
  // ---- 1x1 head (64->10, +bias) + NHWC->NCHW ----
  {
    dim3 g(1048576 / TM, 1);
    k_gemm<<<g, 256, 0, stream>>>(abuf, 64, 64, nullptr, 0, 0, 0, 0, 0, 0, 0, 0, 0, 0,
                                  wt_hd, 64, gbuf, 10, 64, 64, out_b, 10);
    const long long tot = 2048LL * 10 * 32 * 16;
    k_final<<<ewb(tot), 256, 0, stream>>>(gbuf, (float*)d_out, tot);
  }
}
